// SwitchTransformerFFN_56573309224005
// MI455X (gfx1250) — matrix operand order repaired and checked by reading, not compiled, not run on hardware
//
#include <hip/hip_runtime.h>
#include <hip/hip_bf16.h>
#include <math.h>

// ---------------- problem constants ----------------
#define NB 4
#define NS 4096
#define NTOK (NB * NS)               // 16384 tokens
#define DMODEL 1024
#define DFF 2048
#define NEXP 8
#define ROWS 64                      // rows per workgroup tile
#define MAX_PAD (NTOK + ROWS * NEXP) // 16896 padded rows (segments padded to 64)
#define NTILES (MAX_PAD / ROWS)      // 264 row tiles
#define KCHUNK 256                   // K staging chunk for A
#define APITCH (KCHUNK / 2 + 4)      // 132 uints per LDS A row (pad kills bank conflicts)

typedef __attribute__((ext_vector_type(16))) __bf16 v16bf;
typedef __attribute__((ext_vector_type(8)))  float  v8f;
typedef __attribute__((ext_vector_type(4)))  unsigned int u32x4;

union Frag {
    v16bf v;
    unsigned int u[8];
    u32x4 q[2];
};

__device__ __forceinline__ unsigned short f2bf(float f) {
    unsigned int u = __float_as_uint(f);
    unsigned int r = u + 0x7FFFu + ((u >> 16) & 1u);   // round-to-nearest-even
    return (unsigned short)(r >> 16);
}

__device__ __forceinline__ v8f wmma_bf16(const Frag& a, const Frag& b, v8f c) {
    return __builtin_amdgcn_wmma_f32_16x16x32_bf16(
        false, a.v, false, b.v, (short)0, c, false, false);
}

__device__ __forceinline__ float gelu_exact(float v) {
    return 0.5f * v * (1.0f + erff(v * 0.70710678118654752f));
}

// ---------------- 1. init: perm = -1, counts = 0 ----------------
__global__ void moe_init(int* __restrict__ perm, int* __restrict__ counts) {
    int i = blockIdx.x * 256 + threadIdx.x;
    if (i < MAX_PAD) perm[i] = -1;
    if (i < NEXP) counts[i] = 0;
}

// ---------------- 2. weight fp32 -> bf16 with transpose ----------------
// in: [E][R][C] fp32   ->   out: [E][C][R] bf16   (tiled 32x32 through LDS)
__global__ __launch_bounds__(256) void moe_transpose_bf16(
    const float* __restrict__ in, unsigned short* __restrict__ out, int R, int C)
{
    __shared__ unsigned short tile[32][33];
    const int e = blockIdx.z;
    const float* src = in + (size_t)e * R * C;
    unsigned short* dst = out + (size_t)e * R * C;
    const int c0 = blockIdx.x * 32, r0 = blockIdx.y * 32;
    const int tx = threadIdx.x, ty = threadIdx.y;      // block (32, 8)
#pragma unroll
    for (int j = 0; j < 4; ++j) {
        int r = r0 + ty + j * 8;
        tile[ty + j * 8][tx] = f2bf(src[(size_t)r * C + c0 + tx]);
    }
    __syncthreads();
#pragma unroll
    for (int j = 0; j < 4; ++j) {
        int c = c0 + ty + j * 8;
        dst[(size_t)c * R + r0 + tx] = tile[tx][ty + j * 8];
    }
}

// ---------------- 3. router: one wave32 per token (+ x -> bf16) ----------------
__global__ __launch_bounds__(256) void moe_router(
    const float* __restrict__ x, const float* __restrict__ Wr, const float* __restrict__ br,
    unsigned short* __restrict__ xbf, float* __restrict__ wgt,
    int* __restrict__ eidx, int* __restrict__ counts)
{
    int tok  = (blockIdx.x * 256 + threadIdx.x) >> 5;
    int lane = threadIdx.x & 31;
    if (tok >= NTOK) return;

    const float* xr = x + (size_t)tok * DMODEL;
    float acc[NEXP];
#pragma unroll
    for (int e = 0; e < NEXP; ++e) acc[e] = 0.0f;

#pragma unroll
    for (int it = 0; it < DMODEL / 128; ++it) {
        int d = it * 128 + lane * 4;
        float4 xv = *(const float4*)(xr + d);
        unsigned int* xo = (unsigned int*)(xbf + (size_t)tok * DMODEL + d);
        xo[0] = (unsigned)f2bf(xv.x) | ((unsigned)f2bf(xv.y) << 16);
        xo[1] = (unsigned)f2bf(xv.z) | ((unsigned)f2bf(xv.w) << 16);
        float xs[4] = {xv.x, xv.y, xv.z, xv.w};
#pragma unroll
        for (int q = 0; q < 4; ++q) {
            const float* w = Wr + (size_t)(d + q) * NEXP;
#pragma unroll
            for (int e = 0; e < NEXP; ++e) acc[e] += xs[q] * w[e];
        }
    }
#pragma unroll
    for (int e = 0; e < NEXP; ++e)
#pragma unroll
        for (int m = 16; m > 0; m >>= 1)
            acc[e] += __shfl_xor(acc[e], m, 32);

    if (lane == 0) {
        float lmax = -1e30f; int best = 0;
#pragma unroll
        for (int e = 0; e < NEXP; ++e) {
            float l = acc[e] + br[e];
            if (l > lmax) { lmax = l; best = e; }
        }
        float s = 0.0f;
#pragma unroll
        for (int e = 0; e < NEXP; ++e) s += expf((acc[e] + br[e]) - lmax);
        wgt[tok]  = 1.0f / s;
        eidx[tok] = best;
        atomicAdd(&counts[best], 1);
    }
}

// ---------------- 4. padded prefix offsets (pad to ROWS) ----------------
__global__ void moe_offsets(const int* __restrict__ counts, int* __restrict__ offs,
                            int* __restrict__ cursor) {
    if (threadIdx.x == 0) {
        int o = 0;
        offs[0] = 0;
        for (int e = 0; e < NEXP; ++e) {
            cursor[e] = o;
            o += ((counts[e] + ROWS - 1) / ROWS) * ROWS;
            offs[e + 1] = o;
        }
    }
}

// ---------------- 5. scatter tokens into expert-grouped perm ----------------
__global__ void moe_scatter(const int* __restrict__ eidx, int* __restrict__ cursor,
                            int* __restrict__ perm) {
    int t = blockIdx.x * 256 + threadIdx.x;
    if (t >= NTOK) return;
    int pos = atomicAdd(&cursor[eidx[t]], 1);
    perm[pos] = t;
}

// A 16x32 bf16 fragment from LDS chunk: lane L -> row m=L%16,
// halves 0..15 = K kb..kb+15 contiguous, kb=(L<16)?0:16.
__device__ __forceinline__ void load_a_frag(Frag& a, const unsigned int* s_a,
                                            int row, int kk_u, int hi) {
    const unsigned int* p = s_a + row * APITCH + kk_u + hi * 8;
    a.q[0] = *(const u32x4*)p;
    a.q[1] = *(const u32x4*)(p + 4);
}
// B 32x16 bf16 fragment straight from transposed weights in global/L2:
// lane L -> col n=L%16 (row of Wt), K klo..klo+15 contiguous, klo=(L<16)?0:16.
__device__ __forceinline__ void load_b_frag(Frag& b, const unsigned short* colp, int k) {
    const u32x4* q = (const u32x4*)(colp + k);
    b.q[0] = q[0];
    b.q[1] = q[1];
}

// ---------------- 6. GEMM1: h = GELU(gather(x) @ W1[e] + b1[e]) ----------------
// block: 256 thr = 8 waves; tile: 64 rows x 256 cols; wave: 32 cols, 4x2 accumulators
__global__ __launch_bounds__(256) void moe_gemm1(
    const unsigned short* __restrict__ xbf, const unsigned short* __restrict__ w1t,
    const float* __restrict__ b1, const int* __restrict__ perm,
    const int* __restrict__ offs, unsigned short* __restrict__ hbf)
{
    const int t0 = blockIdx.x * ROWS;
    const int nb = blockIdx.y * 256;
    if (perm[t0] < 0) return;                    // inactive tail tile

    __shared__ int s_tok[ROWS];
    __shared__ __align__(16) unsigned int s_a[ROWS * APITCH];

    const int tid = threadIdx.x;
    int e = 0;
#pragma unroll
    for (int q = 1; q < NEXP; ++q) e += (t0 >= offs[q]) ? 1 : 0;
    if (tid < ROWS) s_tok[tid] = perm[t0 + tid];
    __syncthreads();

    const int wave = tid >> 5, lane = tid & 31;
    const int ln16 = lane & 15;
    const int hi   = (lane < 16) ? 0 : 1;
    const int wn   = nb + wave * 32;
    const unsigned short* bcol = w1t + ((size_t)e * DFF + wn + ln16) * DMODEL;

    v8f acc[4][2];
#pragma unroll
    for (int mi = 0; mi < 4; ++mi) { acc[mi][0] = (v8f){}; acc[mi][1] = (v8f){}; }
    const unsigned int* xu = (const unsigned int*)xbf;

    for (int kc = 0; kc < DMODEL; kc += KCHUNK) {
        __syncthreads();
#pragma unroll
        for (int j = 0; j < 32; ++j) {          // stage A chunk: 8192 uints
            int idx = tid + 256 * j;
            int r = idx >> 7, c = idx & 127;
            int tok = s_tok[r];
            unsigned int v = 0;
            if (tok >= 0) v = xu[(size_t)tok * (DMODEL / 2) + (kc >> 1) + c];
            s_a[r * APITCH + c] = v;
        }
        __syncthreads();
        if (kc + KCHUNK < DMODEL)
            __builtin_prefetch(bcol + kc + KCHUNK, 0, 0);
#pragma unroll
        for (int kk = 0; kk < KCHUNK; kk += 32) {
            Frag b0, b1;
            int k = kc + kk + hi * 16;
            load_b_frag(b0, bcol, k);
            load_b_frag(b1, bcol + 16 * DMODEL, k);
#pragma unroll
            for (int mi = 0; mi < 4; ++mi) {
                Frag a;
                load_a_frag(a, s_a, mi * 16 + ln16, kk >> 1, hi);
                acc[mi][0] = wmma_bf16(a, b0, acc[mi][0]);
                acc[mi][1] = wmma_bf16(a, b1, acc[mi][1]);
            }
        }
    }

    const float bias0 = b1[(size_t)e * DFF + wn + ln16];
    const float bias1 = b1[(size_t)e * DFF + wn + 16 + ln16];
#pragma unroll
    for (int mi = 0; mi < 4; ++mi)
#pragma unroll
        for (int i = 0; i < 8; ++i) {
            int m = mi * 16 + i + hi * 8;
            hbf[(size_t)(t0 + m) * DFF + wn + ln16] =
                f2bf(gelu_exact(acc[mi][0][i] + bias0));
            hbf[(size_t)(t0 + m) * DFF + wn + 16 + ln16] =
                f2bf(gelu_exact(acc[mi][1][i] + bias1));
        }
}

// ---------------- 7. GEMM2: out[tok] = (h @ W2[e] + b2[e]) * wgt[tok] ----------------
__global__ __launch_bounds__(256) void moe_gemm2(
    const unsigned short* __restrict__ hbf, const unsigned short* __restrict__ w2t,
    const float* __restrict__ b2, const int* __restrict__ perm,
    const int* __restrict__ offs, const float* __restrict__ wgt,
    float* __restrict__ out)
{
    const int t0 = blockIdx.x * ROWS;
    const int nb = blockIdx.y * 256;
    if (perm[t0] < 0) return;

    __shared__ int s_tok[ROWS];
    __shared__ __align__(16) unsigned int s_a[ROWS * APITCH];

    const int tid = threadIdx.x;
    int e = 0;
#pragma unroll
    for (int q = 1; q < NEXP; ++q) e += (t0 >= offs[q]) ? 1 : 0;
    if (tid < ROWS) s_tok[tid] = perm[t0 + tid];
    __syncthreads();

    const int wave = tid >> 5, lane = tid & 31;
    const int ln16 = lane & 15;
    const int hi   = (lane < 16) ? 0 : 1;
    const int wn   = nb + wave * 32;
    const unsigned short* bcol = w2t + ((size_t)e * DMODEL + wn + ln16) * DFF;

    v8f acc[4][2];
#pragma unroll
    for (int mi = 0; mi < 4; ++mi) { acc[mi][0] = (v8f){}; acc[mi][1] = (v8f){}; }
    const unsigned int* hu = (const unsigned int*)hbf;

    for (int kc = 0; kc < DFF; kc += KCHUNK) {
        __syncthreads();
#pragma unroll
        for (int j = 0; j < 32; ++j) {          // stage A chunk (rows already grouped)
            int idx = tid + 256 * j;
            int r = idx >> 7, c = idx & 127;
            s_a[r * APITCH + c] = hu[(size_t)(t0 + r) * (DFF / 2) + (kc >> 1) + c];
        }
        __syncthreads();
        if (kc + KCHUNK < DFF)
            __builtin_prefetch(bcol + kc + KCHUNK, 0, 0);
#pragma unroll
        for (int kk = 0; kk < KCHUNK; kk += 32) {
            Frag b0, b1;
            int k = kc + kk + hi * 16;
            load_b_frag(b0, bcol, k);
            load_b_frag(b1, bcol + 16 * DFF, k);
#pragma unroll
            for (int mi = 0; mi < 4; ++mi) {
                Frag a;
                load_a_frag(a, s_a, mi * 16 + ln16, kk >> 1, hi);
                acc[mi][0] = wmma_bf16(a, b0, acc[mi][0]);
                acc[mi][1] = wmma_bf16(a, b1, acc[mi][1]);
            }
        }
    }

    const float bias0 = b2[(size_t)e * DMODEL + wn + ln16];
    const float bias1 = b2[(size_t)e * DMODEL + wn + 16 + ln16];
#pragma unroll
    for (int mi = 0; mi < 4; ++mi)
#pragma unroll
        for (int i = 0; i < 8; ++i) {
            int m = mi * 16 + i + hi * 8;
            int tk = s_tok[m];
            if (tk >= 0) {
                float w = wgt[tk];
                out[(size_t)tk * DMODEL + wn + ln16]      = (acc[mi][0][i] + bias0) * w;
                out[(size_t)tk * DMODEL + wn + 16 + ln16] = (acc[mi][1][i] + bias1) * w;
            }
        }
}

// ---------------- host launcher ----------------
extern "C" void kernel_launch(void* const* d_in, const int* in_sizes, int n_in,
                              void* d_out, int out_size, void* d_ws, size_t ws_size,
                              hipStream_t stream)
{
    (void)in_sizes; (void)n_in; (void)out_size; (void)ws_size;
    const float* x  = (const float*)d_in[0];
    const float* Wr = (const float*)d_in[1];
    const float* br = (const float*)d_in[2];
    const float* W1 = (const float*)d_in[3];
    const float* b1 = (const float*)d_in[4];
    const float* W2 = (const float*)d_in[5];
    const float* b2 = (const float*)d_in[6];
    float* out = (float*)d_out;

    char* ws = (char*)d_ws;
    size_t off = 0;
    auto carve = [&](size_t bytes) -> char* {
        char* p = ws + off;
        off = (off + bytes + 255) & ~(size_t)255;
        return p;
    };
    unsigned short* xbf = (unsigned short*)carve((size_t)NTOK * DMODEL * 2);
    unsigned short* w1t = (unsigned short*)carve((size_t)NEXP * DMODEL * DFF * 2); // [E][F][D]
    unsigned short* w2t = (unsigned short*)carve((size_t)NEXP * DFF * DMODEL * 2); // [E][D][F]
    unsigned short* hbf = (unsigned short*)carve((size_t)MAX_PAD * DFF * 2);
    float* wgt    = (float*)carve((size_t)NTOK * 4);
    int*   eidx   = (int*)carve((size_t)NTOK * 4);
    int*   perm   = (int*)carve((size_t)MAX_PAD * 4);
    int*   counts = (int*)carve(NEXP * 4);
    int*   offs   = (int*)carve((NEXP + 1) * 4);
    int*   cursor = (int*)carve(NEXP * 4);

    // 1. init
    moe_init<<<(MAX_PAD + 255) / 256, 256, 0, stream>>>(perm, counts);
    // 2. weights -> bf16, transposed so B fragments are K-contiguous
    moe_transpose_bf16<<<dim3(DFF / 32, DMODEL / 32, NEXP), dim3(32, 8), 0, stream>>>(
        W1, w1t, DMODEL, DFF);
    moe_transpose_bf16<<<dim3(DMODEL / 32, DFF / 32, NEXP), dim3(32, 8), 0, stream>>>(
        W2, w2t, DFF, DMODEL);
    // 3. router (+ x -> bf16)
    moe_router<<<NTOK / 8, 256, 0, stream>>>(x, Wr, br, xbf, wgt, eidx, counts);
    // 4. padded offsets
    moe_offsets<<<1, 32, 0, stream>>>(counts, offs, cursor);
    // 5. scatter
    moe_scatter<<<NTOK / 256, 256, 0, stream>>>(eidx, cursor, perm);
    // 6. GEMM1 (WMMA bf16, GELU epilogue)
    moe_gemm1<<<dim3(NTILES, DFF / 256), 256, 0, stream>>>(xbf, w1t, b1, perm, offs, hbf);
    // 7. GEMM2 (WMMA bf16, weight-scaled scatter store)
    moe_gemm2<<<dim3(NTILES, DMODEL / 256), 256, 0, stream>>>(hbf, w2t, b2, perm, offs, wgt, out);
}
